// GCN_27616639713759
// MI455X (gfx1250) — matrix operand order repaired and checked by reading, not compiled, not run on hardware
//
#include <hip/hip_runtime.h>
#include <stdint.h>

// ---------------------------------------------------------------------------
// Types for CDNA5 WMMA (wave32): v16bf A/B fragments, v8f accumulator.
// ---------------------------------------------------------------------------
typedef __attribute__((ext_vector_type(16))) __bf16 v16bf;
typedef __attribute__((ext_vector_type(8)))  float  v8f;

// 16-byte vector type matching the async-LDS builtin's parameter type
typedef int v4i __attribute__((vector_size(16)));
typedef __attribute__((address_space(1))) v4i* v4i_gptr;   // pointer to global v4i
typedef __attribute__((address_space(3))) v4i* v4i_lptr;   // pointer to LDS v4i

#if __has_builtin(__builtin_amdgcn_global_load_async_to_lds_b128)
#define HAS_ASYNC_LDS 1
#else
#define HAS_ASYNC_LDS 0
#endif

__device__ __forceinline__ void wait_async_zero() {
#if HAS_ASYNC_LDS
#if __has_builtin(__builtin_amdgcn_s_wait_asynccnt)
  __builtin_amdgcn_s_wait_asynccnt(0);
#else
  asm volatile("s_wait_asynccnt 0x0" ::: "memory");
#endif
#endif
}

__device__ __forceinline__ unsigned short f32_to_bf16(float f) {
  unsigned int b = __float_as_uint(f);
  unsigned int r = b + 0x7FFFu + ((b >> 16) & 1u);   // round-to-nearest-even
  return (unsigned short)(r >> 16);
}

// Native no-return f32 atomic add (STOREcnt-tracked, lands in L2 atomic units).
// Forced via inline asm so it can never expand to a CAS loop.
__device__ __forceinline__ void atomic_fadd_agent(float* p, float v) {
  asm volatile("global_atomic_add_f32 %0, %1, off" :: "v"(p), "v"(v) : "memory");
}

// ---------------------------------------------------------------------------
// Elementwise helpers
// ---------------------------------------------------------------------------
// 8 elements per thread: 2x float4 in, 1x uint4 (8 bf16) out
__global__ void cast_bf16_kernel(const float* __restrict__ in,
                                 unsigned short* __restrict__ out,
                                 long long n8) {
  long long i = (long long)blockIdx.x * blockDim.x + threadIdx.x;
  if (i >= n8) return;
  const float4* p = reinterpret_cast<const float4*>(in + i * 8);
  float4 a = p[0], b = p[1];
  union { unsigned short s[8]; uint4 u; } r;
  r.s[0] = f32_to_bf16(a.x); r.s[1] = f32_to_bf16(a.y);
  r.s[2] = f32_to_bf16(a.z); r.s[3] = f32_to_bf16(a.w);
  r.s[4] = f32_to_bf16(b.x); r.s[5] = f32_to_bf16(b.y);
  r.s[6] = f32_to_bf16(b.z); r.s[7] = f32_to_bf16(b.w);
  *reinterpret_cast<uint4*>(out + i * 8) = r.u;
}

__global__ void init_bias_kernel(float* __restrict__ dst,
                                 const float* __restrict__ bias,
                                 long long n, int mask) {
  long long i = (long long)blockIdx.x * blockDim.x + threadIdx.x;
  if (i < n) dst[i] = bias[(int)i & mask];
}

__global__ void relu_ip_kernel(float* __restrict__ h, long long n) {
  long long i = (long long)blockIdx.x * blockDim.x + threadIdx.x;
  if (i < n) h[i] = fmaxf(h[i], 0.0f);
}

__global__ void relu_cast_kernel(const float* __restrict__ in,
                                 unsigned short* __restrict__ out,
                                 long long n) {
  long long i = (long long)blockIdx.x * blockDim.x + threadIdx.x;
  if (i < n) out[i] = f32_to_bf16(fmaxf(in[i], 0.0f));
}

__global__ void zero_kernel(float* __restrict__ p, int n) {
  int i = blockIdx.x * blockDim.x + threadIdx.x;
  if (i < n) p[i] = 0.0f;
}

// ---------------------------------------------------------------------------
// COO SpMM: dst[rows[e], f..f+3] += vals[e] * src[cols[e], f..f+3]
// One thread = one edge x 4 features; indices fetched once per 4 features.
// Gathers/scatters land in the 192 MB L2 (activation matrices are 51-205 MB).
// ---------------------------------------------------------------------------
__global__ void spmm_atomic_kernel(const int* __restrict__ rows,
                                   const int* __restrict__ cols,
                                   const float* __restrict__ vals,
                                   const float* __restrict__ src,
                                   float* __restrict__ dst,
                                   long long total4, int tshift, int tmask,
                                   int F) {
  long long i = (long long)blockIdx.x * blockDim.x + threadIdx.x;
  if (i >= total4) return;
  int e = (int)(i >> tshift);
  int f = ((int)i & tmask) << 2;
  float v = vals[e];
  const float4 s = *reinterpret_cast<const float4*>(&src[(size_t)cols[e] * F + f]);
  float* base = &dst[(size_t)rows[e] * F + f];
  atomic_fadd_agent(base + 0, v * s.x);
  atomic_fadd_agent(base + 1, v * s.y);
  atomic_fadd_agent(base + 2, v * s.z);
  atomic_fadd_agent(base + 3, v * s.w);
}

// ---------------------------------------------------------------------------
// BatchNorm stats over rows for F == 256 columns (thread == column: coalesced)
// stats[c] = sum, stats[256+c] = sum of squares
// ---------------------------------------------------------------------------
__global__ void colstats256_kernel(const float* __restrict__ h,
                                   float* __restrict__ stats, int Nrows) {
  int c  = threadIdx.x;                 // 0..255
  int r0 = blockIdx.x * 256;
  int r1 = min(r0 + 256, Nrows);
  float s = 0.0f, sq = 0.0f;
  for (int r = r0; r < r1; ++r) {
    float v = h[(size_t)r * 256 + c];
    s += v; sq += v * v;
  }
  atomic_fadd_agent(&stats[c], s);
  atomic_fadd_agent(&stats[256 + c], sq);
}

__global__ void bn_apply_cast_kernel(const float* __restrict__ h,
                                     const float* __restrict__ stats,
                                     unsigned short* __restrict__ out_bf16,
                                     long long n, float invN) {
  long long i = (long long)blockIdx.x * blockDim.x + threadIdx.x;
  if (i >= n) return;
  int c = (int)i & 255;
  float m   = stats[c] * invN;
  float var = stats[256 + c] * invN - m * m;
  float v   = (h[i] - m) * rsqrtf(var + 1e-5f);
  out_bf16[i] = f32_to_bf16(v);
}

// ---------------------------------------------------------------------------
// bf16 WMMA GEMM: C(f32, MxNc) = A(bf16, MxK) x B(bf16, KxNc)
// Workgroup tile 128x64, 8 waves (4x2), each wave a 32x32 tile built from
// 2x2 v_wmma_f32_16x16x32_bf16. Double-buffered LDS; A tile staged with the
// CDNA5 async global->LDS path (ASYNCcnt), so the next tile's copy overlaps
// the current tile's WMMAs. B staged transposed so fragment loads are
// contiguous ds_load_b128 (stride 40 = conflict-free banks).
// ---------------------------------------------------------------------------
#define TM 128
#define TN 64
#define TK 32
#define LDA 40
#define LDB 40

__global__ __launch_bounds__(256) void gemm_bf16_wmma(
    const unsigned short* __restrict__ A,
    const unsigned short* __restrict__ B,
    float* __restrict__ C,
    int M, int K, int Nc) {
  __shared__ unsigned short lA[2][TM * LDA];
  __shared__ unsigned short lB[2][TN * LDB];   // transposed: lB[.][n][k]

  const int tid  = threadIdx.x;
  const int lane = tid & 31;
  const int wave = tid >> 5;
  const int wm   = wave >> 1;               // 0..3
  const int wn   = wave & 1;                // 0..1
  const int bm   = blockIdx.x * TM;
  const int bn   = blockIdx.y * TN;
  const int half = lane >> 4;               // 0/1
  const int mr   = lane & 15;

  v8f acc[2][2] = {};

  // ---- tile staging ----
  auto stageA = [&](int k0, int buf) {
#pragma unroll
    for (int cch = 0; cch < 2; ++cch) {
      int chunk = tid * 2 + cch;            // 0..511
      int row   = chunk >> 2;               // 0..127
      int kc    = (chunk & 3) * 8;          // 0,8,16,24
#if HAS_ASYNC_LDS
      // OOB rows read valid workspace memory past A; C stores are row-guarded.
      const unsigned short* gsrc = &A[(size_t)(bm + row) * K + k0 + kc];
      unsigned short* ldst = &lA[buf][row * LDA + kc];
      __builtin_amdgcn_global_load_async_to_lds_b128(
          (v4i_gptr)gsrc, (v4i_lptr)ldst, 0, 0);
#else
      uint4 g = make_uint4(0u, 0u, 0u, 0u);
      if (bm + row < M)
        g = *reinterpret_cast<const uint4*>(&A[(size_t)(bm + row) * K + k0 + kc]);
      *reinterpret_cast<uint4*>(&lA[buf][row * LDA + kc]) = g;
#endif
    }
  };
  auto stageB = [&](int k0, int buf) {
    int krow = tid >> 3;                    // 0..31
    int nc   = (tid & 7) * 8;               // 0..56
    uint4 g = *reinterpret_cast<const uint4*>(&B[(size_t)(k0 + krow) * Nc + bn + nc]);
    union { uint4 u; unsigned short s[8]; } t;
    t.u = g;
#pragma unroll
    for (int e = 0; e < 8; ++e)
      lB[buf][(nc + e) * LDB + krow] = t.s[e];
  };

  const int nk = K / TK;
  stageA(0, 0);
  stageB(0, 0);
  wait_async_zero();
  __syncthreads();

  for (int ki = 0; ki < nk; ++ki) {
    const int buf = ki & 1;
    if (ki + 1 < nk) {                      // prefetch next tile into other buffer
      stageA((ki + 1) * TK, buf ^ 1);
      stageB((ki + 1) * TK, buf ^ 1);
    }

    // ---- fragments per ISA 7.12.2 lane layouts ----
    union F32B { uint4 u[2]; v16bf v; };
    F32B af[2], bfr[2];
#pragma unroll
    for (int i = 0; i < 2; ++i) {
      int m = wm * 32 + i * 16 + mr;
      // A: lanes 0-15 hold K {0..7,16..23}; lanes 16-31 hold K {8..15,24..31}
      af[i].u[0] = *reinterpret_cast<const uint4*>(&lA[buf][m * LDA + half * 8]);
      af[i].u[1] = *reinterpret_cast<const uint4*>(&lA[buf][m * LDA + 16 + half * 8]);
    }
#pragma unroll
    for (int j = 0; j < 2; ++j) {
      int n = wn * 32 + j * 16 + mr;
      // B: lanes 0-15 hold K {0..7,16..23}; lanes 16-31 hold K {8..15,24..31}
      bfr[j].u[0] = *reinterpret_cast<const uint4*>(&lB[buf][n * LDB + half * 8]);
      bfr[j].u[1] = *reinterpret_cast<const uint4*>(&lB[buf][n * LDB + 16 + half * 8]);
    }
#pragma unroll
    for (int i = 0; i < 2; ++i)
#pragma unroll
      for (int j = 0; j < 2; ++j)
        acc[i][j] = __builtin_amdgcn_wmma_f32_16x16x32_bf16(
            false, af[i].v, false, bfr[j].v, (short)0, acc[i][j], false, false);

    wait_async_zero();                      // async writes into buf^1 done
    __syncthreads();                        // visible to all waves
  }

  // ---- store C: VGPR r -> row r (lanes 0-15) / r+8 (lanes 16-31), col = lane&15
#pragma unroll
  for (int i = 0; i < 2; ++i) {
    int rowBase = bm + wm * 32 + i * 16 + half * 8;
#pragma unroll
    for (int j = 0; j < 2; ++j) {
      int col = bn + wn * 32 + j * 16 + mr;
#pragma unroll
      for (int r = 0; r < 8; ++r) {
        int row = rowBase + r;
        if (row < M) C[(size_t)row * Nc + col] = acc[i][j][r];
      }
    }
  }
}

// ---------------------------------------------------------------------------
// Host-side pipeline
// ---------------------------------------------------------------------------
static inline long long cdiv(long long a, long long b) { return (a + b - 1) / b; }

static void run_spmm(const int* rows, const int* cols, const float* vals,
                     const float* src, float* dst, long long E, int F,
                     int log2F, hipStream_t stream) {
  long long total4 = E * (F / 4);
  spmm_atomic_kernel<<<cdiv(total4, 256), 256, 0, stream>>>(
      rows, cols, vals, src, dst, total4, log2F - 2, (F / 4) - 1, F);
}

extern "C" void kernel_launch(void* const* d_in, const int* in_sizes, int n_in,
                              void* d_out, int out_size, void* d_ws, size_t ws_size,
                              hipStream_t stream) {
  const float* x    = (const float*)d_in[0];
  const int*   rows = (const int*)d_in[1];
  const int*   cols = (const int*)d_in[2];
  const float* vals = (const float*)d_in[3];
  const float* w0   = (const float*)d_in[4];
  const float* b0   = (const float*)d_in[5];
  const float* w1   = (const float*)d_in[6];
  const float* b1   = (const float*)d_in[7];
  const float* dw0  = (const float*)d_in[8];
  const float* db0  = (const float*)d_in[9];
  const float* dw1  = (const float*)d_in[10];
  const float* db1  = (const float*)d_in[11];

  const int N = in_sizes[0] / 512;   // 100000
  const int E = in_sizes[1];         // 3200000

  // workspace layout
  char* w = (char*)d_ws;
  unsigned short* Abf = (unsigned short*)w;                      // N*512 bf16 (x, then activations)
  size_t off = (size_t)N * 512 * sizeof(unsigned short);
  float* T = (float*)(w + off);      off += (size_t)N * 512 * sizeof(float);   // GEMM out (pre-SpMM)
  float* H = (float*)(w + off);      off += (size_t)N * 256 * sizeof(float);   // post-SpMM activations
  unsigned short* Wbf = (unsigned short*)(w + off); off += (size_t)256 * 512 * sizeof(unsigned short);
  float* stats = (float*)(w + off);  // 512 floats

  float* x_out = (float*)d_out;                  // N x 128
  float* x_rec = x_out + (size_t)N * 128;        // N x 512
  const float invN = 1.0f / (float)N;

  long long n;

  // ===== encoder layer 0: h = bn(relu(spmm(x@w0) + b0)) =====
  n = (long long)N * 512;
  cast_bf16_kernel<<<cdiv(n/8,256),256,0,stream>>>(x, Abf, n/8);
  cast_bf16_kernel<<<cdiv(512*256/8,256),256,0,stream>>>(w0, Wbf, 512*256/8);
  { dim3 g((unsigned)cdiv(N,TM), 256/TN);
    gemm_bf16_wmma<<<g,256,0,stream>>>(Abf, Wbf, T, N, 512, 256); }
  n = (long long)N * 256;
  init_bias_kernel<<<cdiv(n,256),256,0,stream>>>(H, b0, n, 255);
  run_spmm(rows, cols, vals, T, H, E, 256, 8, stream);
  relu_ip_kernel<<<cdiv(n,256),256,0,stream>>>(H, n);
  zero_kernel<<<2,256,0,stream>>>(stats, 512);
  colstats256_kernel<<<cdiv(N,256),256,0,stream>>>(H, stats, N);
  bn_apply_cast_kernel<<<cdiv(n,256),256,0,stream>>>(H, stats, Abf, n, invN);

  // ===== encoder layer 1: x_out = spmm(h@w1) + b1 =====
  cast_bf16_kernel<<<cdiv(256*128/8,256),256,0,stream>>>(w1, Wbf, 256*128/8);
  { dim3 g((unsigned)cdiv(N,TM), 128/TN);
    gemm_bf16_wmma<<<g,256,0,stream>>>(Abf, Wbf, T, N, 256, 128); }
  n = (long long)N * 128;
  init_bias_kernel<<<cdiv(n,256),256,0,stream>>>(x_out, b1, n, 127);
  run_spmm(rows, cols, vals, T, x_out, E, 128, 7, stream);

  // ===== decoder layer 0: r = bn(relu(spmm(relu(x_out)@dw0) + db0)) =====
  relu_cast_kernel<<<cdiv(n,256),256,0,stream>>>(x_out, Abf, n);
  cast_bf16_kernel<<<cdiv(128*256/8,256),256,0,stream>>>(dw0, Wbf, 128*256/8);
  { dim3 g((unsigned)cdiv(N,TM), 256/TN);
    gemm_bf16_wmma<<<g,256,0,stream>>>(Abf, Wbf, T, N, 128, 256); }
  n = (long long)N * 256;
  init_bias_kernel<<<cdiv(n,256),256,0,stream>>>(H, db0, n, 255);
  run_spmm(rows, cols, vals, T, H, E, 256, 8, stream);
  relu_ip_kernel<<<cdiv(n,256),256,0,stream>>>(H, n);
  zero_kernel<<<2,256,0,stream>>>(stats, 512);
  colstats256_kernel<<<cdiv(N,256),256,0,stream>>>(H, stats, N);
  bn_apply_cast_kernel<<<cdiv(n,256),256,0,stream>>>(H, stats, Abf, n, invN);

  // ===== decoder layer 1: x_rec = spmm(r@dw1) + db1 =====
  cast_bf16_kernel<<<cdiv(256*512/8,256),256,0,stream>>>(dw1, Wbf, 256*512/8);
  { dim3 g((unsigned)cdiv(N,TM), 512/TN);
    gemm_bf16_wmma<<<g,256,0,stream>>>(Abf, Wbf, T, N, 256, 512); }
  n = (long long)N * 512;
  init_bias_kernel<<<cdiv(n,256),256,0,stream>>>(x_rec, db1, n, 511);
  run_spmm(rows, cols, vals, T, x_rec, E, 512, 9, stream);
}
